// GraphWeatherForecaster_62491774157380
// MI455X (gfx1250) — compile-verified, matrix operand order checked by reading, not run on hardware
//
#include <hip/hip_runtime.h>
#include <hip/hip_bf16.h>
#include <stdint.h>
#include <stddef.h>

// ---------------------------------------------------------------------------
// MI455X (gfx1250) GraphWeatherForecaster forward.
// Bandwidth-bound (~0.66 TFLOP vs ~5-8 GB moved per call at 23.3 TB/s), so:
// bf16 activations, fused bias/ReLU/LayerNorm/residual epilogues, weights
// pre-packed to WMMA B-fragment order (L2-resident, ~14 MB << 192 MB L2).
// All matmuls: v_wmma_f32_16x16x32_bf16, wave32, EXEC kept all-1s (no
// divergent branches around WMMA; row/tile indices are clamped instead).
// ---------------------------------------------------------------------------

typedef __bf16 bf16;
typedef __attribute__((ext_vector_type(16))) __bf16 v16bf;
typedef __attribute__((ext_vector_type(8)))  float   v8f;
typedef __attribute__((ext_vector_type(4)))  unsigned u32x4;

union Frag16 { v16bf v; u32x4 u[2]; };

__device__ __forceinline__ bf16 f2b(float f) {
  union { float f; unsigned u; } x; x.f = f;
  unsigned r = x.u + 0x7fffu + ((x.u >> 16) & 1u);   // round-to-nearest-even
  unsigned short h = (unsigned short)(r >> 16);
  return __builtin_bit_cast(bf16, h);
}
__device__ __forceinline__ float b2f(bf16 h) {
  union { unsigned u; float f; } x;
  x.u = ((unsigned)__builtin_bit_cast(unsigned short, h)) << 16;
  return x.f;
}
__device__ __forceinline__ unsigned pack2(float a, float b) {
  unsigned la = (unsigned)__builtin_bit_cast(unsigned short, f2b(a));
  unsigned lb = (unsigned)__builtin_bit_cast(unsigned short, f2b(b));
  return (lb << 16) | la;
}

#define LDSTRIDE 260   // 256 + 4 pad: row stride shifts banks by 4

// ---------------------------------------------------------------------------
// Fused GEMM: out = epilogue(A[M,Kp](bf16) @ Wp + bias)
//   MODE 0: bias + ReLU -> bf16 | MODE 1: bias + LN(g,b)(+bf16 res) -> bf16
//   MODE 2: bias (+f32 res) -> f32
// 256 threads = 8 wave32s; tile 32(M) x 256(N); wave grid 2x4; 4 acc/wave.
// K loop ping-pong double-buffered with scalar-only control flow.
// ---------------------------------------------------------------------------
template<int MODE>
__launch_bounds__(256)
__global__ void gemm_mlp(const bf16* __restrict__ A, int lda, int M,
                         const bf16* __restrict__ Wp, int Kp, int Np,
                         const float* __restrict__ bias, int Nact,
                         const float* __restrict__ gamma, const float* __restrict__ beta,
                         const bf16* __restrict__ resb, const float* __restrict__ resf,
                         int res_ld,
                         bf16* __restrict__ outb, float* __restrict__ outf, int out_ld)
{
  __shared__ float smem[32 * LDSTRIDE];
  const int tid  = threadIdx.x;
  const int wvu  = __builtin_amdgcn_readfirstlane(tid >> 5);  // scalar wave id
  const int lane = tid & 31;
  const int mh   = wvu >> 2;       // 0..1 : 16-row half (scalar)
  const int nq   = wvu & 3;        // 0..3 : 64-col quarter (scalar)
  const int row0 = blockIdx.x * 32;
  const int kb   = (lane >> 4) << 3;          // fragment k-base: 0 or 8
  const int NT   = Np >> 4;
  const int NTm1 = NT - 1;

  // A row, clamped (no divergence; OOB rows never stored)
  int arow = row0 + mh * 16 + (lane & 15);
  if (arow > M - 1) arow = M - 1;
  const bf16* ap = A + (size_t)arow * lda;

  // B tile indices, clamped (duplicate tiles land in never-read LDS columns)
  int nt_i[4];
#pragma unroll
  for (int i = 0; i < 4; ++i) {
    int nt = (nq << 2) + i;
    nt_i[i] = (nt > NTm1) ? NTm1 : nt;
  }

  auto loadA = [&](int k0) {
    Frag16 a;
    const bf16* p = ap + k0 + kb;
    a.u[0] = *reinterpret_cast<const u32x4*>(p);
    a.u[1] = *reinterpret_cast<const u32x4*>(p + 16);
    return a;
  };
  auto loadB = [&](int kt, int i) {
    Frag16 b;
    const bf16* p = Wp + ((size_t)(kt * NT + nt_i[i]) * 32 + lane) * 16;
    b.u[0] = *reinterpret_cast<const u32x4*>(p);
    b.u[1] = *reinterpret_cast<const u32x4*>(p + 8);
    return b;
  };

  v8f acc[4];
#pragma unroll
  for (int i = 0; i < 4; ++i) acc[i] = (v8f)(0.0f);

  // ---- software-pipelined K loop (all branch conditions are scalar) ----
  Frag16 aP, bP[4], aQ, bQ[4];
  aP = loadA(0);
#pragma unroll
  for (int i = 0; i < 4; ++i) bP[i] = loadB(0, i);

  for (int k0 = 0; k0 < Kp; k0 += 64) {
    const bool hasQ = (k0 + 32) < Kp;
    if (hasQ) {
      aQ = loadA(k0 + 32);
#pragma unroll
      for (int i = 0; i < 4; ++i) bQ[i] = loadB((k0 >> 5) + 1, i);
    }
#pragma unroll
    for (int i = 0; i < 4; ++i)
      acc[i] = __builtin_amdgcn_wmma_f32_16x16x32_bf16(
          false, aP.v, false, bP[i].v, (short)0, acc[i], false, false);
    if ((k0 + 64) < Kp) {
      aP = loadA(k0 + 64);
#pragma unroll
      for (int i = 0; i < 4; ++i) bP[i] = loadB((k0 >> 5) + 2, i);
    }
    if (hasQ) {
#pragma unroll
      for (int i = 0; i < 4; ++i)
        acc[i] = __builtin_amdgcn_wmma_f32_16x16x32_bf16(
            false, aQ.v, false, bQ[i].v, (short)0, acc[i], false, false);
    }
  }

  // C/D layout: VGPR j -> (row j, col lane) lanes 0-15; (row j+8, col-16).
  const int rl0 = mh * 16 + ((lane >> 4) << 3);
  const int c0  = (nq << 6) + (lane & 15);
#pragma unroll
  for (int i = 0; i < 4; ++i) {
    const int col = c0 + (i << 4);
    const float bv = (col < Nact) ? bias[col] : 0.0f;
#pragma unroll
    for (int j = 0; j < 8; ++j)
      smem[(rl0 + j) * LDSTRIDE + col] = acc[i][j] + bv;
  }
  __syncthreads();

  if (MODE == 0) {                        // ReLU -> bf16 (packed dword stores)
    const int cu = (tid & 127) << 1;
    for (int r = (tid >> 7); r < 32; r += 2) {
      const int grow = row0 + r;
      if (grow < M && cu < Nact) {
        float v0 = fmaxf(smem[r * LDSTRIDE + cu], 0.0f);
        float v1 = fmaxf(smem[r * LDSTRIDE + cu + 1], 0.0f);
        *reinterpret_cast<unsigned*>(outb + (size_t)grow * out_ld + cu) = pack2(v0, v1);
      }
    }
  } else if (MODE == 1) {                 // LayerNorm (+res) -> bf16
#pragma unroll
    for (int rr = 0; rr < 4; ++rr) {
      const int r = (wvu << 2) + rr;
      const int grow = row0 + r;
      float s = 0.f, s2 = 0.f;
      for (int c = lane << 1; c < Nact; c += 64) {
        float v0 = smem[r * LDSTRIDE + c];
        float v1 = smem[r * LDSTRIDE + c + 1];
        s += v0 + v1; s2 += v0 * v0 + v1 * v1;
      }
#pragma unroll
      for (int off = 16; off > 0; off >>= 1) {
        s  += __shfl_xor(s,  off, 32);
        s2 += __shfl_xor(s2, off, 32);
      }
      const float inv  = 1.0f / (float)Nact;
      const float mu   = s * inv;
      const float var  = s2 * inv - mu * mu;
      const float rstd = rsqrtf(var + 1e-5f);
      if (grow < M) {
        for (int c = lane << 1; c < Nact; c += 64) {
          float v0 = (smem[r * LDSTRIDE + c]     - mu) * rstd * gamma[c]     + beta[c];
          float v1 = (smem[r * LDSTRIDE + c + 1] - mu) * rstd * gamma[c + 1] + beta[c + 1];
          if (resb) {
            unsigned ru = *reinterpret_cast<const unsigned*>(
                resb + (size_t)grow * res_ld + c);
            v0 += b2f(__builtin_bit_cast(bf16, (unsigned short)(ru & 0xffffu)));
            v1 += b2f(__builtin_bit_cast(bf16, (unsigned short)(ru >> 16)));
          }
          *reinterpret_cast<unsigned*>(outb + (size_t)grow * out_ld + c) = pack2(v0, v1);
        }
      }
    }
  } else {                                // plain + f32 residual -> f32
    const int cu = (tid & 127) << 1;
    for (int r = (tid >> 7); r < 32; r += 2) {
      const int grow = row0 + r;
      if (grow < M && cu < Nact) {
        float v0 = smem[r * LDSTRIDE + cu];
        float v1 = smem[r * LDSTRIDE + cu + 1];
        if (resf) {
          v0 += resf[(size_t)grow * res_ld + cu];
          v1 += resf[(size_t)grow * res_ld + cu + 1];
        }
        outf[(size_t)grow * out_ld + cu]     = v0;
        outf[(size_t)grow * out_ld + cu + 1] = v1;
      }
    }
  }
}

// ---------------------------------------------------------------------------
// Repack f32 W[K][N] -> bf16 WMMA B-fragment order, zero-padded to Kp x Np.
// Per (kt,nt) tile: 32 lanes x 16 bf16 contiguous. n = nt*16 + (lane&15);
// k = kt*32 + (v<4 ? kb+2v+lo : 16+kb+2(v-4)+lo), kb = (lane>=16)*8.
// ---------------------------------------------------------------------------
__global__ void repack_w(const float* __restrict__ W, int K, int N,
                         int Kp, int Np, bf16* __restrict__ out)
{
  const int total = (Kp >> 5) * (Np >> 4) * 512;
  const int i = blockIdx.x * 256 + threadIdx.x;
  if (i >= total) return;
  const int e    = i & 15;
  const int lane = (i >> 4) & 31;
  const int tile = i >> 9;
  const int NT = Np >> 4;
  const int nt = tile % NT, kt = tile / NT;
  const int n  = nt * 16 + (lane & 15);
  const int v  = e >> 1, lo = e & 1, kb = (lane >> 4) << 3;
  const int kl = (v < 4) ? (kb + 2 * v + lo) : (16 + kb + 2 * (v - 4) + lo);
  const int k  = kt * 32 + kl;
  const float val = (k < K && n < N) ? W[(size_t)k * N + n] : 0.0f;
  out[i] = f2b(val);
}

// f32 [rows,Kin] -> bf16 [rows,Kout] zero-padded
__global__ void pad_cast(const float* __restrict__ in, int Kin,
                         bf16* __restrict__ out, int Kout, long total)
{
  const long i = (long)blockIdx.x * 256 + threadIdx.x;
  if (i >= total) return;
  const long r = i / Kout;
  const int  c = (int)(i - r * Kout);
  out[i] = (c < Kin) ? f2b(in[r * Kin + c]) : f2b(0.0f);
}

__global__ void zero_u32(unsigned* __restrict__ p, long n) {
  const long i = (long)blockIdx.x * 256 + threadIdx.x;
  if (i < n) p[i] = 0u;
}

__global__ void copy_u32(const unsigned* __restrict__ s, unsigned* __restrict__ d, long n) {
  const long i = (long)blockIdx.x * 256 + threadIdx.x;
  if (i < n) d[i] = s[i];
}

// fill [rows,256] bf16 with one broadcast row z[256]
__global__ void bcast_row(const bf16* __restrict__ z, long rows, bf16* __restrict__ out) {
  const long i = (long)blockIdx.x * 256 + threadIdx.x;       // uint index
  const unsigned* zu = (const unsigned*)z;
  unsigned* ou = (unsigned*)out;
  if (i < rows * 128) ou[i] = zu[i & 127];
}

// out[b*E+e, 0:768] = [xs[b,src[e]] | xd[b,dst[e]] | ebuf[b*E+e]]  (bf16)
__global__ void edge_concat(const bf16* __restrict__ xs, int Ns,
                            const bf16* __restrict__ xd, int Nd,
                            const bf16* __restrict__ e,
                            const int* __restrict__ src, const int* __restrict__ dst,
                            int E, bf16* __restrict__ out)
{
  const int t = threadIdx.x;
  const long r = (long)blockIdx.x * 2 + (t >> 7);
  if (r >= 2L * E) return;
  const int b  = (int)(r / E);
  const int ei = (int)(r - (long)b * E);
  const int ci = t & 127;
  const unsigned* ps = (const unsigned*)(xs + ((size_t)b * Ns + src[ei]) * 256);
  const unsigned* pd = (const unsigned*)(xd + ((size_t)b * Nd + dst[ei]) * 256);
  const unsigned* pe = (const unsigned*)(e + (size_t)r * 256);
  unsigned* po = (unsigned*)(out + (size_t)r * 768);
  po[ci]       = ps[ci];
  po[128 + ci] = pd[ci];
  po[256 + ci] = pe[ci];
}

// out[r, 0:512] = [x[r] | bf16(agg[r])]
__global__ void node_concat(const bf16* __restrict__ x, const float* __restrict__ agg,
                            long rows, bf16* __restrict__ out)
{
  const int t = threadIdx.x;
  const long r = (long)blockIdx.x * 2 + (t >> 7);
  if (r >= rows) return;
  const int ci = t & 127;
  const unsigned* px = (const unsigned*)(x + r * 256);
  unsigned* po = (unsigned*)(out + r * 512);
  po[ci] = px[ci];
  const float a0 = agg[r * 256 + ci * 2];
  const float a1 = agg[r * 256 + ci * 2 + 1];
  po[128 + ci] = pack2(a0, a1);
}

// agg[b*Nn + dst[e]] += f32(enew[b*E+e])   (64 threads/edge, 4 cols each)
__global__ void scatter_add(const bf16* __restrict__ e, const int* __restrict__ dst,
                            int E, int Nn, float* __restrict__ agg)
{
  const long idx = (long)blockIdx.x * 256 + threadIdx.x;   // over 2*E*64
  const int  t  = (int)(idx & 63);
  const long er = idx >> 6;
  if (er >= 2L * E) return;
  const int b  = (int)(er / E);
  const int ei = (int)(er - (long)b * E);
  const long drow = (long)b * Nn + dst[ei];
  const bf16* src = e + er * 256 + t * 4;
  float* dp = agg + drow * 256 + t * 4;
#pragma unroll
  for (int j = 0; j < 4; ++j) atomicAdd(dp + j, b2f(src[j]));
}

// ---------------------------------------------------------------------------
// Host orchestration
// ---------------------------------------------------------------------------
namespace {
struct MlpDesc {
  const float* W[3]; const float* b[3]; const float* g; const float* be;
  int K[3], N[3];
  bf16* Wp[3]; int Kp[3], Np[3];
};
static inline int rup(int x, int m) { return (x + m - 1) / m * m; }
}

extern "C" void kernel_launch(void* const* d_in, const int* in_sizes, int n_in,
                              void* d_out, int out_size, void* d_ws, size_t ws_size,
                              hipStream_t stream)
{
  (void)in_sizes; (void)n_in; (void)out_size; (void)ws_size;
  constexpr int B = 2, NG = 64800, NL = 5882, F = 78, D = 256;
  constexpr int EE = 64800, EP = 35292, ED = 64800, NB = 9, H = 128;
  const int MG = B * NG;
  const int ML = B * NL;

  const float* features   = (const float*)d_in[0];
  const float* enc_eattr  = (const float*)d_in[1];
  const float* proc_eattr = (const float*)d_in[2];
  const float* dec_eattr  = (const float*)d_in[3];
  const int* enc_src  = (const int*)d_in[218];
  const int* enc_dst  = (const int*)d_in[219];
  const int* proc_src = (const int*)d_in[220];
  const int* proc_dst = (const int*)d_in[221];
  const int* dec_src  = (const int*)d_in[222];
  const int* dec_dst  = (const int*)d_in[223];

  auto getp = [&](int j) -> const float* { return (const float*)d_in[4 + j]; };
  auto mkmlp = [&](int base, int K0, int N0, int N1, int N2, bool ln) {
    MlpDesc m{};
    m.W[0] = getp(base + 0); m.W[1] = getp(base + 1); m.W[2] = getp(base + 2);
    m.b[0] = getp(base + 3); m.b[1] = getp(base + 4); m.b[2] = getp(base + 5);
    m.g  = ln ? getp(base + 6) : nullptr;
    m.be = ln ? getp(base + 7) : nullptr;
    m.K[0] = K0; m.K[1] = N0; m.K[2] = N1;
    m.N[0] = N0; m.N[1] = N1; m.N[2] = N2;
    return m;
  };

  // pytree leaf order (sorted keys): dec_block(edge,node), dec_edge, dec_out,
  // enc_block(edge,node), enc_edge, grid_embed, proc_blocks[0..8], proc_edge
  MlpDesc dec_be = mkmlp(0,  3 * D, D, D, D, true);
  MlpDesc dec_bn = mkmlp(8,  2 * D, D, D, D, true);
  MlpDesc dec_e  = mkmlp(16, 3,     D, D, D, true);
  MlpDesc dec_o  = mkmlp(24, D,     H, H, F, false);
  MlpDesc enc_be = mkmlp(30, 3 * D, D, D, D, true);
  MlpDesc enc_bn = mkmlp(38, 2 * D, D, D, D, true);
  MlpDesc enc_e  = mkmlp(46, 3,     D, D, D, true);
  MlpDesc gemb   = mkmlp(54, F,     D, D, D, true);
  MlpDesc pbe[NB], pbn[NB];
  for (int i = 0; i < NB; ++i) {
    pbe[i] = mkmlp(62 + 16 * i,     3 * D, D, D, D, true);
    pbn[i] = mkmlp(62 + 16 * i + 8, 2 * D, D, D, D, true);
  }
  MlpDesc proc_e = mkmlp(206, 3, D, D, D, true);

  char* wcur = (char*)d_ws;
  auto walloc = [&](size_t bytes) -> void* {
    void* p = (void*)wcur;
    wcur += (bytes + 255) & ~(size_t)255;
    return p;
  };

  MlpDesc* all[9 + 2 * NB] = { &dec_be, &dec_bn, &dec_e, &dec_o, &enc_be,
                               &enc_bn, &enc_e, &gemb, &proc_e };
  for (int i = 0; i < NB; ++i) { all[9 + 2 * i] = &pbe[i]; all[10 + 2 * i] = &pbn[i]; }
  for (int mi = 0; mi < 9 + 2 * NB; ++mi) {
    MlpDesc& m = *all[mi];
    for (int l = 0; l < 3; ++l) {
      m.Kp[l] = rup(m.K[l], 32);
      m.Np[l] = rup(m.N[l], 16);
      m.Wp[l] = (bf16*)walloc((size_t)m.Kp[l] * m.Np[l] * sizeof(bf16));
      const int total = (m.Kp[l] >> 5) * (m.Np[l] >> 4) * 512;
      repack_w<<<dim3((total + 255) / 256), dim3(256), 0, stream>>>(
          m.W[l], m.K[l], m.N[l], m.Kp[l], m.Np[l], m.Wp[l]);
    }
  }

  bf16* featbf  = (bf16*)walloc((size_t)MG * 96 * 2);
  bf16* eattrbf = (bf16*)walloc((size_t)EE * 32 * 2);
  bf16* x_grid  = (bf16*)walloc((size_t)MG * D * 2);
  bf16* x_lat   = (bf16*)walloc((size_t)ML * D * 2);
  bf16* e_buf   = (bf16*)walloc((size_t)2 * EE * D * 2);
  bf16* cat     = (bf16*)walloc((size_t)2 * EE * 768 * 2);
  bf16* t0      = (bf16*)walloc((size_t)MG * D * 2);
  float* agg    = (float*)walloc((size_t)MG * D * 4);
  bf16* zin     = (bf16*)walloc(96 * 2);
  bf16* zh0     = (bf16*)walloc(D * 2);
  bf16* zh1     = (bf16*)walloc(D * 2);
  bf16* zrow    = (bf16*)walloc(D * 2);
  bf16* t1      = cat;   // alias: 2nd hidden buffer lives in the concat region

  auto gemm = [&](int mode, const bf16* A, int lda, int M,
                  const bf16* Wp, int Kp, int Np, const float* bias, int Nact,
                  const float* g, const float* be,
                  const bf16* resb, const float* resf, int res_ld,
                  bf16* outb, float* outf, int out_ld) {
    dim3 grid((unsigned)((M + 31) / 32)), block(256);
    switch (mode) {
      case 0: gemm_mlp<0><<<grid, block, 0, stream>>>(A, lda, M, Wp, Kp, Np, bias,
                 Nact, g, be, resb, resf, res_ld, outb, outf, out_ld); break;
      case 1: gemm_mlp<1><<<grid, block, 0, stream>>>(A, lda, M, Wp, Kp, Np, bias,
                 Nact, g, be, resb, resf, res_ld, outb, outf, out_ld); break;
      default: gemm_mlp<2><<<grid, block, 0, stream>>>(A, lda, M, Wp, Kp, Np, bias,
                 Nact, g, be, resb, resf, res_ld, outb, outf, out_ld); break;
    }
  };

  auto run_mlp = [&](MlpDesc& m, const bf16* A, int lda, int M, int finalMode,
                     const bf16* resb, const float* resf, int res_ld,
                     bf16* outb, float* outf, int out_ld, bf16* h0, bf16* h1) {
    gemm(0, A,  lda,    M, m.Wp[0], m.Kp[0], m.Np[0], m.b[0], m.N[0],
         nullptr, nullptr, nullptr, nullptr, 0, h0, nullptr, m.N[0]);
    gemm(0, h0, m.N[0], M, m.Wp[1], m.Kp[1], m.Np[1], m.b[1], m.N[1],
         nullptr, nullptr, nullptr, nullptr, 0, h1, nullptr, m.N[1]);
    gemm(finalMode, h1, m.N[1], M, m.Wp[2], m.Kp[2], m.Np[2], m.b[2], m.N[2],
         m.g, m.be, resb, resf, res_ld, outb, outf, out_ld);
  };

  auto zero_f32 = [&](float* p, long n) {
    zero_u32<<<dim3((unsigned)((n + 255) / 256)), dim3(256), 0, stream>>>((unsigned*)p, n);
  };

  auto run_block = [&](MlpDesc& me, MlpDesc& mn, const bf16* xs, int Ns,
                       bf16* xd, int Nd, bf16* e, int E,
                       const int* src, const int* dst) {
    const long erows = 2L * E;
    edge_concat<<<dim3((unsigned)((erows + 1) / 2)), dim3(256), 0, stream>>>(
        xs, Ns, xd, Nd, e, src, dst, E, cat);
    run_mlp(me, cat, 768, (int)erows, 1, e, nullptr, D, e, nullptr, D, t0, t1);
    zero_f32(agg, 2L * Nd * D);
    const long st = 2L * E * 64;
    scatter_add<<<dim3((unsigned)((st + 255) / 256)), dim3(256), 0, stream>>>(
        e, dst, E, Nd, agg);
    const long nrows = 2L * Nd;
    node_concat<<<dim3((unsigned)((nrows + 1) / 2)), dim3(256), 0, stream>>>(
        xd, agg, nrows, cat);
    run_mlp(mn, cat, 512, (int)nrows, 1, xd, nullptr, D, xd, nullptr, D, t0, t1);
  };

  auto run_embed = [&](MlpDesc& m, const float* eattr, int E) {
    pad_cast<<<dim3((unsigned)(((long)E * 32 + 255) / 256)), dim3(256), 0, stream>>>(
        eattr, 3, eattrbf, 32, (long)E * 32);
    run_mlp(m, eattrbf, 32, E, 1, nullptr, nullptr, 0, e_buf, nullptr, D, t0, t1);
    const long nu = (long)E * 128;
    copy_u32<<<dim3((unsigned)((nu + 255) / 256)), dim3(256), 0, stream>>>(
        (const unsigned*)e_buf, (unsigned*)(e_buf + (size_t)E * D), nu);
  };

  // ---------------- Encoder ----------------
  pad_cast<<<dim3((unsigned)(((long)MG * 96 + 255) / 256)), dim3(256), 0, stream>>>(
      features, F, featbf, 96, (long)MG * 96);
  run_mlp(gemb, featbf, 96, MG, 1, nullptr, nullptr, 0, x_grid, nullptr, D, t0, t1);

  zero_u32<<<dim3(1), dim3(256), 0, stream>>>((unsigned*)zin, 48);
  run_mlp(gemb, zin, 96, 1, 1, nullptr, nullptr, 0, zrow, nullptr, D, zh0, zh1);
  bcast_row<<<dim3((unsigned)(((long)ML * 128 + 255) / 256)), dim3(256), 0, stream>>>(
      zrow, (long)ML, x_lat);

  run_embed(enc_e, enc_eattr, EE);
  run_block(enc_be, enc_bn, x_grid, NG, x_lat, NL, e_buf, EE, enc_src, enc_dst);

  // ---------------- Processor ----------------
  run_embed(proc_e, proc_eattr, EP);
  for (int i = 0; i < NB; ++i)
    run_block(pbe[i], pbn[i], x_lat, NL, x_lat, NL, e_buf, EP, proc_src, proc_dst);

  // ---------------- Decoder ----------------
  run_embed(dec_e, dec_eattr, ED);
  run_block(dec_be, dec_bn, x_lat, NL, x_grid, NG, e_buf, ED, dec_src, dec_dst);

  run_mlp(dec_o, x_grid, D, MG, 2, nullptr, features, F, nullptr, (float*)d_out, F,
          t0, t1);
}
